// SlidingWindowAttention_2138893714156
// MI455X (gfx1250) — matrix operand order repaired and checked by reading, not compiled, not run on hardware
//
#include <hip/hip_runtime.h>

// ---------------------------------------------------------------------------
// Sliding-window attention on gfx1250 (MI455X).
// All matmuls via v_wmma_f32_16x16x32_bf16; global->LDS staging via
// CDNA5 async copies (global_load_async_to_lds_b128 / s_wait_asynccnt);
// software-pipelined LDS fragment loads; vectorized b128 epilogues.
// B=2 S=8192 D=1024 H=16 dk=dv=64 W=128.
// ---------------------------------------------------------------------------

typedef __attribute__((ext_vector_type(16))) __bf16 v16bf;
typedef __attribute__((ext_vector_type(8)))  float  v8f;

union FragBF {
  v16bf v;
  uint4 u[2];
};

__device__ __forceinline__ unsigned short f2bf(float f) {
  unsigned int u = __builtin_bit_cast(unsigned int, f);
  u += 0x7FFFu + ((u >> 16) & 1u);      // round-to-nearest-even
  return (unsigned short)(u >> 16);
}
__device__ __forceinline__ float bf2f(unsigned short h) {
  unsigned int u = ((unsigned int)h) << 16;
  return __builtin_bit_cast(float, u);
}

// CDNA5 async global->LDS 16B copy (ASYNCcnt-tracked, no VGPR bounce).
// Generic pointer to a __shared__ object: low 32 bits are the LDS byte offset.
__device__ __forceinline__ void async_g2l_b128(void* lds, const void* g) {
  unsigned loff = (unsigned)(size_t)lds;
  asm volatile("global_load_async_to_lds_b128 %0, %1, off"
               :: "v"(loff), "v"(g) : "memory");
}
__device__ __forceinline__ void wait_async() {
  asm volatile("s_wait_asynccnt 0" ::: "memory");
}

// ---------------------------------------------------------------------------
// f32 -> bf16 bulk convert (n multiple of 4)
// ---------------------------------------------------------------------------
__global__ __launch_bounds__(256) void cvt_bf16_kernel(
    const float* __restrict__ src, unsigned short* __restrict__ dst, int n) {
  int i = (blockIdx.x * 256 + threadIdx.x) * 4;
  if (i >= n) return;
  float4 v = *(const float4*)(src + i);
  uint2 o;
  o.x = (unsigned)f2bf(v.x) | ((unsigned)f2bf(v.y) << 16);
  o.y = (unsigned)f2bf(v.z) | ((unsigned)f2bf(v.w) << 16);
  *(uint2*)(dst + i) = o;
}

// ---------------------------------------------------------------------------
// f32 [1024,1024] -> bf16 transposed [n][k] (tiled LDS transpose, one-time)
// ---------------------------------------------------------------------------
__global__ __launch_bounds__(256) void cvt_bf16_transpose_kernel(
    const float* __restrict__ src, unsigned short* __restrict__ dst) {
  __shared__ float tile[32][33];
  const int bx = blockIdx.x * 32;                 // n tile
  const int by = blockIdx.y * 32;                 // k tile
  const int tx = threadIdx.x & 31, ty = threadIdx.x >> 5;  // 32x8
#pragma unroll
  for (int i = 0; i < 32; i += 8)
    tile[ty + i][tx] = src[(size_t)(by + ty + i) * 1024 + bx + tx];
  __syncthreads();
#pragma unroll
  for (int i = 0; i < 32; i += 8)
    dst[(size_t)(bx + ty + i) * 1024 + by + tx] = f2bf(tile[tx][ty + i]);
}

// ---------------------------------------------------------------------------
// Tiled bf16 GEMM: C[M,1024] = A[M,1024] * W[1024,1024] (W pre-transposed).
// 128x128 block tile, K-step 32, 256 threads = 8 waves, 8 wmma accums/wave.
// Double-buffered LDS fed by async global->LDS b128 copies; B-fragment loads
// software-pipelined one tile ahead of each WMMA.
// mode 0: z selects (Wt,O) triple; output bf16 [B,H,S,64] via LDS-bounced
//         contiguous b128 stores.  mode 1: f32 row-major to outF.
// ---------------------------------------------------------------------------
__global__ __launch_bounds__(256) void gemm_bf16(
    const unsigned short* __restrict__ A,     // [M,1024] bf16
    const unsigned short* __restrict__ Wt0,   // [1024,1024] bf16, [n][k]
    const unsigned short* __restrict__ Wt1,
    const unsigned short* __restrict__ Wt2,
    unsigned short* __restrict__ O0,
    unsigned short* __restrict__ O1,
    unsigned short* __restrict__ O2,
    float* __restrict__ outF,
    int mode) {
  const int K = 1024;
  __shared__ unsigned short As[2][128 * 32];   // [m][k]  2 x 8 KB
  __shared__ unsigned short Bs[2][128 * 32];   // [n][k]  2 x 8 KB
  __shared__ unsigned short Cs[128 * 128];     // epilogue bounce, 32 KB

  const unsigned short* Wt = (blockIdx.z == 0) ? Wt0 : (blockIdx.z == 1 ? Wt1 : Wt2);
  unsigned short* O = (blockIdx.z == 0) ? O0 : (blockIdx.z == 1 ? O1 : O2);

  const int t     = threadIdx.x;
  const int wave  = t >> 5;
  const int lane  = t & 31;
  const int lhalf = lane >> 4;
  const int l16   = lane & 15;

  const int m0 = blockIdx.y * 128;
  const int n0 = blockIdx.x * 128;

  v8f acc[8];
#pragma unroll
  for (int j = 0; j < 8; ++j)
#pragma unroll
    for (int e = 0; e < 8; ++e) acc[j][e] = 0.f;

  // Stage one 128x32 A tile and 128x32 B tile (each 8 KB) as b128 async
  // copies: 512 uint4 per tile, 2 per thread per tile.
  auto stage = [&](int kt, int buf) {
#pragma unroll
    for (int i = 0; i < 2; ++i) {
      const int u   = i * 256 + t;      // uint4 index in tile
      const int row = u >> 2;           // 0..127
      const int c   = (u & 3) * 8;      // bf16 col 0/8/16/24
      async_g2l_b128(&As[buf][row * 32 + c], A  + (size_t)(m0 + row) * K + kt + c);
      async_g2l_b128(&Bs[buf][row * 32 + c], Wt + (size_t)(n0 + row) * K + kt + c);
    }
  };

  stage(0, 0);
  wait_async();
  __syncthreads();

  const int kbB = lhalf * 16;                   // B-frag: contiguous 16 K
#pragma unroll 1
  for (int kt = 0; kt < K; kt += 32) {
    const int buf = (kt >> 5) & 1;
    if (kt + 32 < K) stage(kt + 32, buf ^ 1);   // overlap with WMMAs below

    FragBF af, bcur;
    const int am  = wave * 16 + l16;
    const int kbA = lhalf * 16;                 // A-frag: contiguous 16 K
    af.u[0] = *(const uint4*)(&As[buf][am * 32 + kbA]);
    af.u[1] = *(const uint4*)(&As[buf][am * 32 + kbA + 8]);
    bcur.u[0] = *(const uint4*)(&Bs[buf][l16 * 32 + kbB]);
    bcur.u[1] = *(const uint4*)(&Bs[buf][l16 * 32 + kbB + 8]);
#pragma unroll
    for (int j = 0; j < 8; ++j) {
      FragBF bnxt;
      if (j < 7) {                              // pipeline next B-frag
        const int bn = (j + 1) * 16 + l16;
        bnxt.u[0] = *(const uint4*)(&Bs[buf][bn * 32 + kbB]);
        bnxt.u[1] = *(const uint4*)(&Bs[buf][bn * 32 + kbB + 8]);
      }
      acc[j] = __builtin_amdgcn_wmma_f32_16x16x32_bf16(
          false, af.v, false, bcur.v, (short)0, acc[j], false, false);
      if (j < 7) bcur = bnxt;
    }
    wait_async();
    __syncthreads();
  }

  if (mode == 0) {
    // bounce C tile through LDS, then contiguous 128B rows to [B,H,S,64]
#pragma unroll
    for (int j = 0; j < 8; ++j)
#pragma unroll
      for (int r = 0; r < 8; ++r)
        Cs[(wave * 16 + lhalf * 8 + r) * 128 + j * 16 + l16] = f2bf(acc[j][r]);
    __syncthreads();
    const int h2  = t >> 7;          // which of the 2 heads in this tile
    const int row = t & 127;
    const int m   = m0 + row;
    const int bb  = m >> 13;         // / 8192
    const int ss  = m & 8191;
    const int h0  = n0 >> 6;
    unsigned short* dst = O + ((size_t)(bb * 16 + h0 + h2) * 8192 + ss) * 64;
    const unsigned short* srcp = Cs + row * 128 + h2 * 64;
#pragma unroll
    for (int i = 0; i < 8; ++i)
      ((uint4*)dst)[i] = ((const uint4*)srcp)[i];
  } else {
#pragma unroll
    for (int j = 0; j < 8; ++j)
#pragma unroll
      for (int r = 0; r < 8; ++r) {
        const int m  = m0 + wave * 16 + lhalf * 8 + r;
        const int nn = n0 + j * 16 + l16;
        outF[(size_t)m * 1024 + nn] = acc[j][r];
      }
  }
}

// ---------------------------------------------------------------------------
// Banded attention: one block per (query-block n, head h, batch b).
// K tile async-staged to LDS, V staged transposed; scores 128x384 via WMMA
// -> masked bf16 LDS; row softmax; P*V via WMMA; ctx LDS-bounced to
// contiguous b128 stores in bf16 [B,S,H*64].
// ---------------------------------------------------------------------------
__global__ __launch_bounds__(256) void swa_attn(
    const unsigned short* __restrict__ Q,
    const unsigned short* __restrict__ Km,
    const unsigned short* __restrict__ V,
    unsigned short* __restrict__ Ctx) {
  const int S = 8192;
  __shared__ unsigned short sc[128 * 384];  // scores/probs/ctx   96 KB
  __shared__ unsigned short Ks[384 * 64];   // K  [key][kk]       48 KB
  __shared__ unsigned short Vt[64 * 384];   // V^T [d][key]       48 KB

  const int n = blockIdx.x, h = blockIdx.y, b = blockIdx.z;
  const int t     = threadIdx.x;
  const int wave  = t >> 5;
  const int lane  = t & 31;
  const int lhalf = lane >> 4;
  const int l16   = lane & 15;

  const size_t headOff = (size_t)(b * 16 + h) * S * 64;
  const unsigned short* Qh = Q  + headOff;
  const unsigned short* Kh = Km + headOff;
  const unsigned short* Vh = V  + headOff;

  // Stage K (async b128, layout already B-frag friendly) and V transposed.
  // 384 rows x 64 bf16 = 3072 uint4 each; clamped rows (masked probs are 0).
  for (int u = t; u < 3072; u += 256) {
    const int row = u >> 3;          // key row 0..383
    const int c   = (u & 7) * 8;     // bf16 col
    int jg = n * 128 - 128 + row;
    jg = jg < 0 ? 0 : (jg >= S ? S - 1 : jg);
    async_g2l_b128(&Ks[row * 64 + c], Kh + (size_t)jg * 64 + c);
    uint4 vv = *(const uint4*)(Vh + (size_t)jg * 64 + c);
    const unsigned short* hp = (const unsigned short*)&vv;
#pragma unroll
    for (int i = 0; i < 8; ++i) Vt[(c + i) * 384 + row] = hp[i];
  }

  // Q fragments (2 K-steps of 32 over dk=64) held in registers
  FragBF qf[2];
  {
    const int qs = n * 128 + wave * 16 + l16;
    const unsigned short* qp = Qh + (size_t)qs * 64;
#pragma unroll
    for (int ks = 0; ks < 2; ++ks) {
      const int kb = ks * 32 + lhalf * 16;
      qf[ks].u[0] = *(const uint4*)(qp + kb);
      qf[ks].u[1] = *(const uint4*)(qp + kb + 8);
    }
  }
  wait_async();
  __syncthreads();

  const float scale = 0.125f;   // 1/sqrt(64)
#pragma unroll 1
  for (int jt = 0; jt < 24; ++jt) {
    v8f acc;
#pragma unroll
    for (int e = 0; e < 8; ++e) acc[e] = 0.f;

    const int jl = jt * 16 + l16;               // key column 0..383
    const int jg = n * 128 - 128 + jl;          // global key index
    const unsigned short* kp = &Ks[jl * 64];
    FragBF bf0, bf1;                            // both K-frags up front
    bf0.u[0] = *(const uint4*)(kp + lhalf * 16);
    bf0.u[1] = *(const uint4*)(kp + lhalf * 16 + 8);
    bf1.u[0] = *(const uint4*)(kp + 32 + lhalf * 16);
    bf1.u[1] = *(const uint4*)(kp + 32 + lhalf * 16 + 8);
    acc = __builtin_amdgcn_wmma_f32_16x16x32_bf16(
        false, qf[0].v, false, bf0.v, (short)0, acc, false, false);
    acc = __builtin_amdgcn_wmma_f32_16x16x32_bf16(
        false, qf[1].v, false, bf1.v, (short)0, acc, false, false);
#pragma unroll
    for (int r = 0; r < 8; ++r) {
      const int iloc = wave * 16 + lhalf * 8 + r;
      const int diff = jl - 128 - iloc;
      const bool band  = (diff <= 128) && (diff >= -128);
      const bool valid = (jg >= 0) && (jg < S);
      const float sv = (band && valid) ? acc[r] * scale : -1e9f;
      sc[iloc * 384 + jl] = f2bf(sv);
    }
  }
  __syncthreads();

  // row softmax (threads 0..127, one row each)
  if (t < 128) {
    unsigned short* row = sc + t * 384;
    float mx = -3.4e38f;
    for (int j = 0; j < 384; ++j) {
      const float v = bf2f(row[j]);
      mx = v > mx ? v : mx;
    }
    float sum = 0.f;
    for (int j = 0; j < 384; ++j) sum += __expf(bf2f(row[j]) - mx);
    const float inv = 1.f / sum;
    for (int j = 0; j < 384; ++j)
      row[j] = f2bf(__expf(bf2f(row[j]) - mx) * inv);
  }
  __syncthreads();

  // ctx = P[128,384] * V[384,64], V-frag loads pipelined ahead of WMMAs
  v8f cacc[4];
#pragma unroll
  for (int c = 0; c < 4; ++c)
#pragma unroll
    for (int e = 0; e < 8; ++e) cacc[c][e] = 0.f;

#pragma unroll 1
  for (int ks = 0; ks < 12; ++ks) {
    FragBF pf, vcur;
    const int prow = wave * 16 + l16;
    const int kbA  = ks * 32 + lhalf * 16;
    const int kbB  = ks * 32 + lhalf * 16;
    pf.u[0] = *(const uint4*)(sc + prow * 384 + kbA);
    pf.u[1] = *(const uint4*)(sc + prow * 384 + kbA + 8);
    vcur.u[0] = *(const uint4*)(Vt + l16 * 384 + kbB);
    vcur.u[1] = *(const uint4*)(Vt + l16 * 384 + kbB + 8);
#pragma unroll
    for (int c = 0; c < 4; ++c) {
      FragBF vnxt;
      if (c < 3) {
        const int d = (c + 1) * 16 + l16;
        vnxt.u[0] = *(const uint4*)(Vt + d * 384 + kbB);
        vnxt.u[1] = *(const uint4*)(Vt + d * 384 + kbB + 8);
      }
      cacc[c] = __builtin_amdgcn_wmma_f32_16x16x32_bf16(
          false, pf.v, false, vcur.v, (short)0, cacc[c], false, false);
      if (c < 3) vcur = vnxt;
    }
  }

  // bounce ctx tile (128x64) through sc, then contiguous 64B segments out
  __syncthreads();
#pragma unroll
  for (int c = 0; c < 4; ++c)
#pragma unroll
    for (int r = 0; r < 8; ++r)
      sc[(wave * 16 + lhalf * 8 + r) * 64 + c * 16 + l16] = f2bf(cacc[c][r]);
  __syncthreads();
  {
    const int row    = t >> 1;
    const int half32 = (t & 1) * 32;
    const int s      = n * 128 + row;
    unsigned short* dst = Ctx + (size_t)(b * 8192 + s) * 1024 + h * 64 + half32;
    const unsigned short* srcp = sc + row * 64 + half32;
#pragma unroll
    for (int i = 0; i < 4; ++i)
      ((uint4*)dst)[i] = ((const uint4*)srcp)[i];
  }
}

// ---------------------------------------------------------------------------
// Host launcher
// ---------------------------------------------------------------------------
extern "C" void kernel_launch(void* const* d_in, const int* in_sizes, int n_in,
                              void* d_out, int out_size, void* d_ws, size_t ws_size,
                              hipStream_t stream) {
  const float* x  = (const float*)d_in[0];
  const float* Wq = (const float*)d_in[1];
  const float* Wk = (const float*)d_in[2];
  const float* Wv = (const float*)d_in[3];
  const float* Wo = (const float*)d_in[4];
  float* out = (float*)d_out;

  const int nX = 2 * 8192 * 1024;   // 16,777,216 (also = B*H*S*64)
  const int nW = 1024 * 1024;

  char* ws = (char*)d_ws;
  size_t off = 0;
  auto take = [&](size_t elems) -> unsigned short* {
    unsigned short* p = (unsigned short*)(ws + off);
    off += ((elems * 2 + 255) / 256) * 256;
    return p;
  };
  unsigned short* xb  = take(nX);
  unsigned short* wqb = take(nW);   // transposed [n][k]
  unsigned short* wkb = take(nW);
  unsigned short* wvb = take(nW);
  unsigned short* wob = take(nW);
  unsigned short* Qb  = take(nX);   // [B,H,S,64]
  unsigned short* Kb  = take(nX);
  unsigned short* Vb  = take(nX);
  unsigned short* Cx  = take(nX);   // [B,S,1024]
  (void)ws_size; (void)in_sizes; (void)n_in; (void)out_size;

  cvt_bf16_kernel<<<nX / 1024, 256, 0, stream>>>(x, xb, nX);
  cvt_bf16_transpose_kernel<<<dim3(32, 32), 256, 0, stream>>>(Wq, wqb);
  cvt_bf16_transpose_kernel<<<dim3(32, 32), 256, 0, stream>>>(Wk, wkb);
  cvt_bf16_transpose_kernel<<<dim3(32, 32), 256, 0, stream>>>(Wv, wvb);
  cvt_bf16_transpose_kernel<<<dim3(32, 32), 256, 0, stream>>>(Wo, wob);

  // Q/K/V projections (z selects weight/output)
  gemm_bf16<<<dim3(8, 128, 3), 256, 0, stream>>>(
      xb, wqb, wkb, wvb, Qb, Kb, Vb, nullptr, 0);

  // banded attention: grid (nb=64, H=16, B=2)
  swa_attn<<<dim3(64, 16, 2), 256, 0, stream>>>(Qb, Kb, Vb, Cx);

  // output projection -> f32 d_out
  gemm_bf16<<<dim3(8, 128, 1), 256, 0, stream>>>(
      Cx, wob, wob, wob, nullptr, nullptr, nullptr, out, 1);
}
